// AttentionBlock_13245679141132
// MI455X (gfx1250) — matrix operand order repaired and checked by reading, not compiled, not run on hardware
//
#include <hip/hip_runtime.h>

typedef __bf16 bf16_t;
typedef __attribute__((ext_vector_type(8)))  bf16_t v8bf;
typedef __attribute__((ext_vector_type(16))) bf16_t v16bf;
typedef __attribute__((ext_vector_type(8)))  float  v8f;

#define B_   32
#define C_   512
#define N_   1024
#define CPG  16
#define ATT_SCALE 0.04419417382415922f  // 512^-0.5

// workspace element offsets (bf16 units)
#define WQ_OFF 0
#define WK_OFF (512*512)
#define WV_OFF (2*512*512)
#define WP_OFF (3*512*512)
#define HT_OFF (4*512*512)
#define QT_OFF (HT_OFF + B_*N_*C_)
#define KT_OFF (QT_OFF + B_*N_*C_)
#define VV_OFF (KT_OFF + B_*N_*C_)
#define OT_OFF (VV_OFF + B_*N_*C_)

__device__ __forceinline__ v8f wmma_bf16(v16bf a, v16bf b, v8f c) {
  return __builtin_amdgcn_wmma_f32_16x16x32_bf16(false, a, false, b, (short)0, c,
                                                 false, false);
}

// ---- CDNA5 async global->LDS copy (ASYNCcnt-tracked), 16 bytes per lane.
// Flat->LDS address mapping truncates to addr[31:0], so the low 32 bits of a
// generic pointer to __shared__ are the LDS address the VDST operand expects.
__device__ __forceinline__ void async_b128(void* lds, const void* gmem) {
  const unsigned int       l = (unsigned int)(unsigned long long)lds;
  const unsigned long long g = (unsigned long long)gmem;
  asm volatile("global_load_async_to_lds_b128 %0, %1, off"
               :: "v"(l), "v"(g)
               : "memory");
}
__device__ __forceinline__ void wait_async_le4() {
  asm volatile("s_wait_asynccnt 0x4" ::: "memory");
}
__device__ __forceinline__ void wait_async_0() {
  asm volatile("s_wait_asynccnt 0x0" ::: "memory");
}

// A fragment 16x32 (MxK) from row-major [M][K] storage (ld in elements).
// lane: m = lane%16, h = lane/16; elements e: K = (e/8)*16 + h*8 + (e%8)
// Works for global or LDS source (address space inferred after inlining).
__device__ __forceinline__ v16bf frag_a(const bf16_t* base, int ld, int lane) {
  const int m = lane & 15, h = lane >> 4;
  const bf16_t* p = base + (size_t)m * ld;
  const v8bf lo = *reinterpret_cast<const v8bf*>(p + h * 8);
  const v8bf hi = *reinterpret_cast<const v8bf*>(p + 16 + h * 8);
  v16bf r;
#pragma unroll
  for (int e = 0; e < 8; ++e) { r[e] = lo[e]; r[e + 8] = hi[e]; }
  return r;
}

// B fragment 32x16 (KxN) from row-major [N][K] storage.
// lane: n = lane%16, h = lane/16; elements e: K = (e/8)*16 + h*8 + (e%8)
__device__ __forceinline__ v16bf frag_b(const bf16_t* base, int ld, int lane) {
  const int n = lane & 15, h = lane >> 4;
  const bf16_t* p = base + (size_t)n * ld;
  const v8bf lo = *reinterpret_cast<const v8bf*>(p + h * 8);
  const v8bf hi = *reinterpret_cast<const v8bf*>(p + 16 + h * 8);
  v16bf r;
#pragma unroll
  for (int e = 0; e < 8; ++e) { r[e] = lo[e]; r[e + 8] = hi[e]; }
  return r;
}

// cooperative async stage of one 128x32 A tile + 128x32 B tile (bf16)
__device__ __forceinline__ void stage_tiles(bf16_t (*dA)[32], const bf16_t* gA,
                                            bf16_t (*dB)[32], const bf16_t* gB,
                                            int ldg, int t) {
#pragma unroll
  for (int s = 0; s < 2; ++s) {
    const int c   = t + s * 256;      // 0..511 chunks of 16B
    const int row = c >> 2;
    const int col = (c & 3) * 8;
    async_b128(&dA[row][col], gA + (size_t)row * ldg + col);
    async_b128(&dB[row][col], gB + (size_t)row * ldg + col);
  }
}

// ---------------------------------------------------------------- weights
__global__ void convert_weights(const float* __restrict__ qw, const float* __restrict__ kw,
                                const float* __restrict__ vw, const float* __restrict__ pw,
                                bf16_t* __restrict__ ws) {
  const int i = blockIdx.x * 256 + threadIdx.x;   // 0 .. 262143
  ws[WQ_OFF + i] = (bf16_t)qw[i];
  ws[WK_OFF + i] = (bf16_t)kw[i];
  ws[WV_OFF + i] = (bf16_t)vw[i];
  ws[WP_OFF + i] = (bf16_t)pw[i];
}

// ---------------------------------------------------------------- groupnorm
// one block per (batch, group); writes hT[b][n][c] bf16 (transposed via LDS)
__global__ void groupnorm_kernel(const float* __restrict__ x, const float* __restrict__ nw,
                                 const float* __restrict__ nb, bf16_t* __restrict__ hT) {
  __shared__ float rs[256], rq[256];
  __shared__ float tile[16][17];
  const int t  = threadIdx.x;
  const int b  = blockIdx.x >> 5;
  const int g  = blockIdx.x & 31;
  const int c0 = g * CPG;
  const float* xb = x + ((size_t)b * C_ + c0) * N_;

  float s1 = 0.f, s2 = 0.f;
#pragma unroll 4
  for (int it = 0; it < 64; ++it) {
    const int idx = t + (it << 8);
    const int cc = idx >> 10, n = idx & 1023;
    const float v = xb[(size_t)cc * N_ + n];
    s1 += v; s2 += v * v;
  }
  rs[t] = s1; rq[t] = s2; __syncthreads();
  for (int off = 128; off > 0; off >>= 1) {
    if (t < off) { rs[t] += rs[t + off]; rq[t] += rq[t + off]; }
    __syncthreads();
  }
  const float mean = rs[0] * (1.f / 16384.f);
  const float var  = rq[0] * (1.f / 16384.f) - mean * mean;
  const float rstd = rsqrtf(var + 1e-5f);

  const int cr = t >> 4, nc = t & 15;
  const float gw = nw[c0 + cr] * rstd;
  const float gb = nb[c0 + cr] - mean * gw;
  bf16_t* ho = hT + (size_t)b * N_ * C_;
  for (int nt = 0; nt < 64; ++nt) {
    const int n0 = nt << 4;
    tile[cr][nc] = xb[(size_t)cr * N_ + n0 + nc] * gw + gb;
    __syncthreads();
    ho[(size_t)(n0 + cr) * C_ + c0 + nc] = (bf16_t)tile[nc][cr];
    __syncthreads();
  }
}

// ---------------------------------------------------------------- QKV GEMM
// Y[o,n] = sum_c W[o,c] * hT[n,c] + bias[o]
// mode 0 -> qT[n][o] * scale ; mode 1 -> kT[n][o] ; mode 2 -> v[o][n]
// A/B tiles double-buffered in LDS via async global->LDS DMA.
__global__ void qkv_gemm(bf16_t* __restrict__ ws, const float* __restrict__ qb,
                         const float* __restrict__ kb, const float* __restrict__ vb) {
  __shared__ bf16_t As[2][128][32];
  __shared__ bf16_t Bs[2][128][32];

  const int bz = blockIdx.z;
  const int b = bz / 3, mode = bz % 3;
  const int t = threadIdx.x;
  const int lane = t & 31, wave = t >> 5;
  const int wm = (wave >> 2) * 64;           // wave m-offset in 128 tile
  const int wn = (wave & 3) * 32;            // wave n-offset in 128 tile
  const int m_off = blockIdx.y * 128 + wm;
  const int n_off = blockIdx.x * 128 + wn;
  const bf16_t* W    = ws + (mode == 0 ? WQ_OFF : (mode == 1 ? WK_OFF : WV_OFF));
  const float*  bias = (mode == 0 ? qb : (mode == 1 ? kb : vb));
  const bf16_t* hT   = ws + HT_OFF + (size_t)b * N_ * C_;
  const bf16_t* gA = W  + (size_t)(blockIdx.y * 128) * C_;
  const bf16_t* gB = hT + (size_t)(blockIdx.x * 128) * C_;

  stage_tiles(As[0], gA, Bs[0], gB, C_, t);  // prologue: kc = 0

  v8f acc[4][2] = {};
  int buf = 0;
  for (int kc = 0; kc < C_; kc += 32, buf ^= 1) {
    __syncthreads();                         // prev readers of buf^1 done
    const bool more = (kc + 32) < C_;
    if (more) stage_tiles(As[buf ^ 1], gA + kc + 32, Bs[buf ^ 1], gB + kc + 32, C_, t);
    if (more) wait_async_le4(); else wait_async_0();
    __syncthreads();                         // tile buf fully in LDS

    v16bf af[4];
#pragma unroll
    for (int i = 0; i < 4; ++i) af[i] = frag_a(&As[buf][wm + 16 * i][0], 32, lane);
    v16bf bfg[2];
#pragma unroll
    for (int j = 0; j < 2; ++j) bfg[j] = frag_b(&Bs[buf][wn + 16 * j][0], 32, lane);
#pragma unroll
    for (int i = 0; i < 4; ++i)
#pragma unroll
      for (int j = 0; j < 2; ++j)
        acc[i][j] = wmma_bf16(af[i], bfg[j], acc[i][j]);
  }

  const int h = lane >> 4, nl = lane & 15;
  if (mode < 2) {
    bf16_t* dst = ws + (mode == 0 ? QT_OFF : KT_OFF) + (size_t)b * N_ * C_;
    const float scl = (mode == 0) ? ATT_SCALE : 1.0f;
#pragma unroll
    for (int i = 0; i < 4; ++i) {
      const int ob = m_off + 16 * i + 8 * h;
      float bia[8];
#pragma unroll
      for (int r = 0; r < 8; ++r) bia[r] = bias[ob + r];
#pragma unroll
      for (int j = 0; j < 2; ++j) {
        const int n = n_off + 16 * j + nl;
        v8bf o;
#pragma unroll
        for (int r = 0; r < 8; ++r) o[r] = (bf16_t)((acc[i][j][r] + bia[r]) * scl);
        *reinterpret_cast<v8bf*>(dst + (size_t)n * C_ + ob) = o;
      }
    }
  } else {
    bf16_t* dst = ws + VV_OFF + (size_t)b * C_ * N_;
#pragma unroll
    for (int i = 0; i < 4; ++i) {
      const int ob = m_off + 16 * i + 8 * h;
      float bia[8];
#pragma unroll
      for (int r = 0; r < 8; ++r) bia[r] = bias[ob + r];
#pragma unroll
      for (int j = 0; j < 2; ++j) {
        const int n = n_off + 16 * j + nl;
#pragma unroll
        for (int r = 0; r < 8; ++r)
          dst[(size_t)(ob + r) * N_ + n] = (bf16_t)(acc[i][j][r] + bia[r]);
      }
    }
  }
}

// ---------------------------------------------------------------- flash attention
// one block per (batch, 32-query tile); keys streamed in 64-wide tiles
__global__ void flash_attn(bf16_t* __restrict__ ws) {
  __shared__ float  S[32][64];
  __shared__ bf16_t P[32][64];
  __shared__ float  red[32][8];
  __shared__ float  mrow[32], lrow[32], arow[32];

  const int b  = blockIdx.y;
  const int i0 = blockIdx.x * 32;
  const int t  = threadIdx.x;
  const int lane = t & 31, wave = t >> 5;
  const bf16_t* qT = ws + QT_OFF + (size_t)b * N_ * C_;
  const bf16_t* kT = ws + KT_OFF + (size_t)b * N_ * C_;
  const bf16_t* V  = ws + VV_OFF + (size_t)b * C_ * N_;
  bf16_t*       oT = ws + OT_OFF + (size_t)b * N_ * C_;

  if (t < 32) { mrow[t] = -3.0e38f; lrow[t] = 0.f; }
  __syncthreads();

  const int sw_i = (wave >> 2) << 4;   // S-tile wave grid: 2(i) x 4(j)
  const int sw_j = (wave & 3) << 4;
  const int i_w  = (wave & 1) << 4;    // out-tile wave grid: 2(i) x 4(c)
  const int c_w  = (wave >> 1) << 7;
  const int h = lane >> 4, nl = lane & 15;
  const int row = t >> 3, sub = t & 7;

  v8f acc[8] = {};

  for (int jt = 0; jt < 16; ++jt) {
    const int j0 = jt << 6;

    // ---- S = (q*scale)^T k  (32 x 64 tile)
    v8f s = {};
    for (int kc = 0; kc < C_; kc += 32) {
      v16bf a  = frag_a(qT + (size_t)(i0 + sw_i) * C_ + kc, C_, lane);
      v16bf bk = frag_b(kT + (size_t)(j0 + sw_j) * C_ + kc, C_, lane);
      s = wmma_bf16(a, bk, s);
    }
#pragma unroll
    for (int r = 0; r < 8; ++r) S[sw_i + 8 * h + r][sw_j + nl] = s[r];
    __syncthreads();

    // ---- online softmax (8 threads per row)
    float mx = -3.0e38f;
#pragma unroll
    for (int e = 0; e < 8; ++e) mx = fmaxf(mx, S[row][sub * 8 + e]);
    red[row][sub] = mx;
    __syncthreads();
    if (sub == 0) {
      float mt = red[row][0];
#pragma unroll
      for (int e = 1; e < 8; ++e) mt = fmaxf(mt, red[row][e]);
      const float mo = mrow[row];
      const float mn = fmaxf(mo, mt);
      arow[row] = __expf(mo - mn);
      mrow[row] = mn;
    }
    __syncthreads();
    const float mn = mrow[row];
    float ps = 0.f;
#pragma unroll
    for (int e = 0; e < 8; ++e) {
      const float p = __expf(S[row][sub * 8 + e] - mn);
      P[row][sub * 8 + e] = (bf16_t)p;
      ps += p;
    }
    red[row][sub] = ps;
    __syncthreads();
    if (sub == 0) {
      float sm = 0.f;
#pragma unroll
      for (int e = 0; e < 8; ++e) sm += red[row][e];
      lrow[row] = arow[row] * lrow[row] + sm;
    }

    // ---- rescale accumulators by alpha[i]
    float al[8];
#pragma unroll
    for (int r = 0; r < 8; ++r) al[r] = arow[i_w + 8 * h + r];
#pragma unroll
    for (int f = 0; f < 8; ++f)
#pragma unroll
      for (int r = 0; r < 8; ++r) acc[f][r] *= al[r];

    // ---- acc += P @ V^T  (P from LDS as A-fragments via ds_load_b128)
#pragma unroll
    for (int ks = 0; ks < 64; ks += 32) {
      v16bf pa = frag_a(&P[i_w][ks], 64, lane);
#pragma unroll
      for (int f = 0; f < 8; ++f) {
        v16bf vb = frag_b(V + (size_t)(c_w + 16 * f) * N_ + j0 + ks, N_, lane);
        acc[f] = wmma_bf16(pa, vb, acc[f]);
      }
    }
    __syncthreads();
  }

  // ---- finalize: divide by l, store oT[n][c] bf16
  float linv[8];
#pragma unroll
  for (int r = 0; r < 8; ++r) linv[r] = 1.f / lrow[i_w + 8 * h + r];
#pragma unroll
  for (int f = 0; f < 8; ++f) {
    const int c = c_w + 16 * f + nl;
#pragma unroll
    for (int r = 0; r < 8; ++r) {
      const int i = i_w + 8 * h + r;
      oT[(size_t)(i0 + i) * C_ + c] = (bf16_t)(acc[f][r] * linv[r]);
    }
  }
}

// ---------------------------------------------------------------- proj GEMM + residual
__global__ void proj_gemm(bf16_t* __restrict__ ws, const float* __restrict__ x,
                          const float* __restrict__ pb, float* __restrict__ out) {
  __shared__ bf16_t As[2][128][32];
  __shared__ bf16_t Bs[2][128][32];

  const int b = blockIdx.z;
  const int t = threadIdx.x;
  const int lane = t & 31, wave = t >> 5;
  const int wm = (wave >> 2) * 64;
  const int wn = (wave & 3) * 32;
  const int m_off = blockIdx.y * 128 + wm;
  const int n_off = blockIdx.x * 128 + wn;
  const bf16_t* oT = ws + OT_OFF + (size_t)b * N_ * C_;
  const bf16_t* gA = ws + WP_OFF + (size_t)(blockIdx.y * 128) * C_;
  const bf16_t* gB = oT + (size_t)(blockIdx.x * 128) * C_;

  stage_tiles(As[0], gA, Bs[0], gB, C_, t);

  v8f acc[4][2] = {};
  int buf = 0;
  for (int kc = 0; kc < C_; kc += 32, buf ^= 1) {
    __syncthreads();
    const bool more = (kc + 32) < C_;
    if (more) stage_tiles(As[buf ^ 1], gA + kc + 32, Bs[buf ^ 1], gB + kc + 32, C_, t);
    if (more) wait_async_le4(); else wait_async_0();
    __syncthreads();

    v16bf af[4];
#pragma unroll
    for (int i = 0; i < 4; ++i) af[i] = frag_a(&As[buf][wm + 16 * i][0], 32, lane);
    v16bf bfg[2];
#pragma unroll
    for (int j = 0; j < 2; ++j) bfg[j] = frag_b(&Bs[buf][wn + 16 * j][0], 32, lane);
#pragma unroll
    for (int i = 0; i < 4; ++i)
#pragma unroll
      for (int j = 0; j < 2; ++j)
        acc[i][j] = wmma_bf16(af[i], bfg[j], acc[i][j]);
  }

  const int h = lane >> 4, nl = lane & 15;
#pragma unroll
  for (int i = 0; i < 4; ++i) {
    const int ob = m_off + 16 * i + 8 * h;
    float bia[8];
#pragma unroll
    for (int r = 0; r < 8; ++r) bia[r] = pb[ob + r];
#pragma unroll
    for (int j = 0; j < 2; ++j) {
      const int n = n_off + 16 * j + nl;
#pragma unroll
      for (int r = 0; r < 8; ++r) {
        const size_t idx = ((size_t)b * C_ + ob + r) * N_ + n;
        out[idx] = x[idx] + bia[r] + acc[i][j][r];
      }
    }
  }
}

extern "C" void kernel_launch(void* const* d_in, const int* in_sizes, int n_in,
                              void* d_out, int out_size, void* d_ws, size_t ws_size,
                              hipStream_t stream) {
  (void)in_sizes; (void)n_in; (void)out_size; (void)ws_size;
  const float* x  = (const float*)d_in[0];
  const float* nw = (const float*)d_in[1];
  const float* nb = (const float*)d_in[2];
  const float* qw = (const float*)d_in[3];
  const float* qb = (const float*)d_in[4];
  const float* kw = (const float*)d_in[5];
  const float* kb = (const float*)d_in[6];
  const float* vw = (const float*)d_in[7];
  const float* vb = (const float*)d_in[8];
  const float* pw = (const float*)d_in[9];
  const float* pb = (const float*)d_in[10];
  float* out = (float*)d_out;
  bf16_t* wsb = (bf16_t*)d_ws;

  convert_weights<<<1024, 256, 0, stream>>>(qw, kw, vw, pw, wsb);
  groupnorm_kernel<<<B_ * 32, 256, 0, stream>>>(x, nw, nb, wsb + HT_OFF);
  qkv_gemm<<<dim3(8, 4, B_ * 3), 256, 0, stream>>>(wsb, qb, kb, vb);
  flash_attn<<<dim3(32, B_), 256, 0, stream>>>(wsb);
  proj_gemm<<<dim3(8, 4, B_), 256, 0, stream>>>(wsb, x, pb, out);
}
